// FalseNegativeContrastiveLoss_32117765439628
// MI455X (gfx1250) — matrix operand order repaired and checked by reading, not compiled, not run on hardware
//
#include <hip/hip_runtime.h>

// FalseNegativeContrastiveLoss — fused WMMA GEMM + epilogue for gfx1250 (MI455X).
//   K0 convert_zero : V,T fp32 -> f16 in ws; zero row/col stat arrays
//   K1 fused_sim    : 128x128 sim tiles; A/B tiles DMA'd into LDS with
//                     global_load_async_to_lds_b128 (double-buffered, ASYNCcnt),
//                     fragments via ds_load_b128, v_wmma_f32_16x16x32_f16 math,
//                     fused epilogue accumulating rowA=sum exp(.5 s),
//                     rowB=sum keep*exp(1.5 s), colA, colB, diag(sim_ii)
//   K2 finalize     : loss = sum_i [-2*s_ii + log((N-1)B_r/(A_r-e05_ii)+eps)
//                                            + log((N-1)B_c/(A_c-e05_ii)+eps)] / (2N)

#define NN 4096
#define DD 512

typedef __attribute__((ext_vector_type(8)))  _Float16 v8h;
typedef __attribute__((ext_vector_type(16))) _Float16 v16h;
typedef __attribute__((ext_vector_type(8)))  float    v8f;

union ABFrag { v16h v; v8h h[2]; };

__global__ __launch_bounds__(256)
void fnc_convert_zero(const float* __restrict__ V, const float* __restrict__ T,
                      _Float16* __restrict__ Vh, _Float16* __restrict__ Th,
                      float* __restrict__ rowA, float* __restrict__ rowB,
                      float* __restrict__ colA, float* __restrict__ colB)
{
    const int idx    = blockIdx.x * blockDim.x + threadIdx.x;
    const int stride = gridDim.x * blockDim.x;
    const int total  = NN * DD;
    for (int i = idx; i < total; i += stride) {
        Vh[i] = (_Float16)V[i];
        Th[i] = (_Float16)T[i];
    }
    if (idx < NN) {
        rowA[idx] = 0.f; rowB[idx] = 0.f;
        colA[idx] = 0.f; colB[idx] = 0.f;
    }
}

// Copy 32 bytes global->LDS as two async b128 DMAs. The 24-bit IOFFSET is added
// to BOTH the LDS and global address (ISA 08_async_tensor §4.4), so offset:16
// fetches the second 16B chunk of the same 32B span.
__device__ __forceinline__ void async_copy32(unsigned lds_addr, const _Float16* gptr)
{
    asm volatile("global_load_async_to_lds_b128 %0, %1, off\n\t"
                 "global_load_async_to_lds_b128 %0, %1, off offset:16"
                 :: "v"(lds_addr), "v"(gptr) : "memory");
}

__device__ __forceinline__ void wait_async_all()
{
    asm volatile("s_wait_asynccnt 0x0" ::: "memory");
}

__global__ __launch_bounds__(256)
void fnc_fused_sim(const _Float16* __restrict__ Vh, const _Float16* __restrict__ Th,
                   const unsigned char* __restrict__ fn_mask,
                   float* __restrict__ rowA, float* __restrict__ rowB,
                   float* __restrict__ colA, float* __restrict__ colB,
                   float* __restrict__ diag)
{
    constexpr float INV_TEMP = 1.0f / 0.07f;

    // double-buffered staging tiles: 128 rows x 32 K, f16 (8 KB each)
    __shared__ __align__(16) _Float16 sA[2][128 * 32];
    __shared__ __align__(16) _Float16 sB[2][128 * 32];
    __shared__ float sRowA[128], sRowB[128], sColA[128], sColB[128];

    const int tid = threadIdx.x;
    if (tid < 128) { sRowA[tid] = 0.f; sRowB[tid] = 0.f; sColA[tid] = 0.f; sColB[tid] = 0.f; }

    const int wave = tid >> 5;          // 8 waves (wave32)
    const int lane = tid & 31;
    const int lr   = lane & 15;         // index within 16-wide tile dim
    const int hl   = lane >> 4;         // half-wave select

    const int blockRow = blockIdx.y * 128;
    const int blockCol = blockIdx.x * 128;
    const int waveRowL = (wave >> 1) * 32;             // local: 2 row tiles of 16
    const int waveColL = (wave & 1) * 64;              // local: 4 col tiles of 16
    const int waveRow  = blockRow + waveRowL;
    const int waveCol  = blockCol + waveColL;

    // staging assignment: thread -> (row, 16-element K chunk)
    const int srow = tid >> 1;
    const int schk = tid & 1;
    const _Float16* gArow = Vh + (size_t)(blockRow + srow) * DD + 16 * schk;
    const _Float16* gBrow = Th + (size_t)(blockCol + srow) * DD + 16 * schk;
    const int sOff = srow * 32 + 16 * schk;

    v8f acc[2][4];
#pragma unroll
    for (int i = 0; i < 2; ++i)
#pragma unroll
        for (int j = 0; j < 4; ++j)
#pragma unroll
            for (int e = 0; e < 8; ++e) acc[i][j][e] = 0.f;

    // prologue: DMA K-slab 0 into buffer 0
    async_copy32((unsigned)(size_t)&sA[0][sOff], gArow);
    async_copy32((unsigned)(size_t)&sB[0][sOff], gBrow);

    const int KSTEPS = DD / 32;
    for (int step = 0; step < KSTEPS; ++step) {
        const int cur = step & 1;
        wait_async_all();        // this thread's DMAs into `cur` are done
        __syncthreads();         // => every thread's DMAs done; prev buffer fully read

        if (step + 1 < KSTEPS) { // DMA next K-slab into the other buffer
            const int kk = (step + 1) * 32;
            async_copy32((unsigned)(size_t)&sA[cur ^ 1][sOff], gArow + kk);
            async_copy32((unsigned)(size_t)&sB[cur ^ 1][sOff], gBrow + kk);
        }

        ABFrag a[2], b[4];
#pragma unroll
        for (int i = 0; i < 2; ++i) {
            // 16-bit A layout: lanes 0-15 hold K=0..7 & 16..23, lanes 16-31 K=8..15 & 24..31
            const _Float16* pa = &sA[cur][(waveRowL + i * 16 + lr) * 32];
            a[i].h[0] = *(const v8h*)(pa + 8 * hl);
            a[i].h[1] = *(const v8h*)(pa + 16 + 8 * hl);
        }
#pragma unroll
        for (int j = 0; j < 4; ++j) {
            // 16-bit B layout: lanes 0-15 hold K=0..7 & 16..23 of col n, lanes 16-31 K=8..15 & 24..31
            const _Float16* pb = &sB[cur][(waveColL + j * 16 + lr) * 32 + 8 * hl];
            b[j].h[0] = *(const v8h*)(pb);
            b[j].h[1] = *(const v8h*)(pb + 16);
        }
#pragma unroll
        for (int i = 0; i < 2; ++i)
#pragma unroll
            for (int j = 0; j < 4; ++j)
                acc[i][j] = __builtin_amdgcn_wmma_f32_16x16x32_f16(
                    false, a[i].v, false, b[j].v, (short)0, acc[i][j], false, false);
    }

    // ---- fused epilogue ----
    float colAccA[4], colAccB[4];
#pragma unroll
    for (int j = 0; j < 4; ++j) { colAccA[j] = 0.f; colAccB[j] = 0.f; }

#pragma unroll
    for (int i = 0; i < 2; ++i) {
        const int rbase = waveRow + i * 16 + 8 * hl;   // C layout: row = e + 8*hl
#pragma unroll
        for (int e = 0; e < 8; ++e) {
            const int gr = rbase + e;
            float rA = 0.f, rB = 0.f;
#pragma unroll
            for (int j = 0; j < 4; ++j) {
                const int gc    = waveCol + j * 16 + lr;
                const float sim = acc[i][j][e] * INV_TEMP;
                const float e05 = __expf(0.5f * sim);
                const bool keep = (gr != gc) && (fn_mask[(size_t)gr * NN + gc] == 0);
                const float e15 = keep ? (e05 * e05 * e05) : 0.f;  // exp(1.5*sim)
                if (gr == gc) diag[gr] = sim;
                rA += e05; rB += e15;
                colAccA[j] += e05; colAccB[j] += e15;
            }
            // reduce row partials across the 16 lanes of this half-wave
#pragma unroll
            for (int m = 1; m < 16; m <<= 1) {
                rA += __shfl_xor(rA, m, 32);
                rB += __shfl_xor(rB, m, 32);
            }
            if (lr == 0) {
                atomicAdd(&sRowA[gr - blockRow], rA);
                atomicAdd(&sRowB[gr - blockRow], rB);
            }
        }
    }
#pragma unroll
    for (int j = 0; j < 4; ++j) {
        // lanes l and l^16 hold the same column (different row halves)
        float cA = colAccA[j] + __shfl_xor(colAccA[j], 16, 32);
        float cB = colAccB[j] + __shfl_xor(colAccB[j], 16, 32);
        if (hl == 0) {
            const int lc = waveColL + j * 16 + lr;
            atomicAdd(&sColA[lc], cA);
            atomicAdd(&sColB[lc], cB);
        }
    }
    __syncthreads();

    if (tid < 128) {
        atomicAdd(&rowA[blockRow + tid], sRowA[tid]);
        atomicAdd(&rowB[blockRow + tid], sRowB[tid]);
        atomicAdd(&colA[blockCol + tid], sColA[tid]);
        atomicAdd(&colB[blockCol + tid], sColB[tid]);
    }
}

__global__ __launch_bounds__(256)
void fnc_finalize(const float* __restrict__ rowA, const float* __restrict__ rowB,
                  const float* __restrict__ colA, const float* __restrict__ colB,
                  const float* __restrict__ diag, float* __restrict__ out)
{
    __shared__ float red[256];
    const int tid = threadIdx.x;
    float s = 0.f;
    for (int i = tid; i < NN; i += 256) {
        const float sim_ii = diag[i];
        const float d05    = __expf(0.5f * sim_ii);
        const float rden   = (float)(NN - 1) * rowB[i] / (rowA[i] - d05);
        const float cden   = (float)(NN - 1) * colB[i] / (colA[i] - d05);
        s += -2.f * sim_ii + __logf(rden + 1e-8f) + __logf(cden + 1e-8f);
    }
    red[tid] = s;
    __syncthreads();
    for (int off = 128; off > 0; off >>= 1) {
        if (tid < off) red[tid] += red[tid + off];
        __syncthreads();
    }
    if (tid == 0) out[0] = red[0] / (2.0f * (float)NN);
}

extern "C" void kernel_launch(void* const* d_in, const int* in_sizes, int n_in,
                              void* d_out, int out_size, void* d_ws, size_t ws_size,
                              hipStream_t stream)
{
    (void)in_sizes; (void)n_in; (void)out_size; (void)ws_size;

    const float*         V    = (const float*)d_in[0];
    const float*         T    = (const float*)d_in[1];
    const unsigned char* mask = (const unsigned char*)d_in[2];
    float*               out  = (float*)d_out;

    // workspace layout: Vh (4MB) | Th (4MB) | rowA | rowB | colA | colB | diag
    char*     ws   = (char*)d_ws;
    _Float16* Vh   = (_Float16*)(ws);
    _Float16* Th   = (_Float16*)(ws + (size_t)NN * DD * sizeof(_Float16));
    float*    rowA = (float*)(ws + 2 * (size_t)NN * DD * sizeof(_Float16));
    float*    rowB = rowA + NN;
    float*    colA = rowB + NN;
    float*    colB = colA + NN;
    float*    diag = colB + NN;

    fnc_convert_zero<<<2048, 256, 0, stream>>>(V, T, Vh, Th, rowA, rowB, colA, colB);

    dim3 grid(NN / 128, NN / 128, 1);
    fnc_fused_sim<<<grid, 256, 0, stream>>>(Vh, Th, mask, rowA, rowB, colA, colB, diag);

    fnc_finalize<<<1, 256, 0, stream>>>(rowA, rowB, colA, colB, diag, out);
}
